// MultiScaleRetention_58548994179831
// MI455X (gfx1250) — compile-verified, matrix operand order checked by reading, not run on hardware
//
#include <hip/hip_runtime.h>
#include <stdint.h>

typedef unsigned short u16;
typedef __attribute__((ext_vector_type(16))) __bf16 v16bf;
typedef __attribute__((ext_vector_type(8)))  float  v8f;

#define B_  2
#define S_  2048
#define E_  2048
#define H_  16
#define HD_ 128

__device__ __forceinline__ u16 f2bf(float f) {
  union { float f; uint32_t u; } a; a.f = f;
  uint32_t r = (a.u + 0x7FFFu + ((a.u >> 16) & 1u)) >> 16;  // RNE
  return (u16)r;
}
__device__ __forceinline__ float bf2f(u16 h) {
  union { uint32_t u; float f; } a; a.u = ((uint32_t)h) << 16;
  return a.f;
}

// Generic LDS pointer -> 32-bit LDS byte address (flat LDS addr keeps the
// LDS offset in the low 32 bits; upper bits are the shared aperture).
__device__ __forceinline__ uint32_t lds_off(const void* p) {
  return (uint32_t)(uintptr_t)p;
}

// CDNA5 async copy global->LDS, 16B per lane, tracked by ASYNCcnt.
__device__ __forceinline__ void async_b128(uint32_t lds, const void* gptr) {
  asm volatile("global_load_async_to_lds_b128 %0, %1, off"
               :: "v"(lds), "v"(gptr) : "memory");
}
__device__ __forceinline__ void wait_async0() {
#if __has_builtin(__builtin_amdgcn_s_wait_asynccnt)
  __builtin_amdgcn_s_wait_asynccnt(0);
#else
  asm volatile("s_wait_asynccnt 0" ::: "memory");
#endif
}

// A-style bf16 fragment (16 rows in lanes, 32 K-values in regs) from a
// row-major [rows][stride] LDS tile. Lanes 0-15 carry K = kbase+{0..7,16..23},
// lanes 16-31 carry K = kbase+{8..15,24..31} (CDNA5 16-bit A layout).
// The B operand of D = A*B uses the same layout applied to B^T.
__device__ __forceinline__ v16bf load_frag(const u16* tile, int stride, int row,
                                           int kbase, int lane) {
  const int r = row + (lane & 15);
  const int half8 = ((lane >> 4) & 1) * 8;
  const u16* p = tile + r * stride + kbase + half8;
  union { v16bf v; uint4 u[2]; } f;
  f.u[0] = *(const uint4*)(p);
  f.u[1] = *(const uint4*)(p + 16);
  return f.v;
}

// ---------------- fp32 -> bf16 conversion (vectorized) ----------------
__global__ void k_cvt(const float* __restrict__ src, u16* __restrict__ dst, int n4) {
  int i = blockIdx.x * blockDim.x + threadIdx.x;
  if (i < n4) {
    const float4 f = ((const float4*)src)[i];
    ushort4 o;
    o.x = f2bf(f.x); o.y = f2bf(f.y); o.z = f2bf(f.z); o.w = f2bf(f.w);
    ((ushort4*)dst)[i] = o;
  }
}

// W [K][N] fp32 (stored [in,out]) -> WT [N][K] bf16, LDS-tiled transpose
__global__ __launch_bounds__(256) void k_cvt_t(const float* __restrict__ W,
                                               u16* __restrict__ WT) {
  __shared__ float t[32][33];
  const int c0 = blockIdx.x * 32, r0 = blockIdx.y * 32;
  const int tx = threadIdx.x, ty = threadIdx.y;
#pragma unroll
  for (int i = 0; i < 32; i += 8)
    t[ty + i][tx] = W[(size_t)(r0 + ty + i) * E_ + c0 + tx];   // coalesced read
  __syncthreads();
#pragma unroll
  for (int i = 0; i < 32; i += 8)
    WT[(size_t)(c0 + ty + i) * E_ + r0 + tx] = f2bf(t[tx][ty + i]);  // coalesced write
}

// ---------------- bf16 WMMA GEMM: C = act(A @ WT^T + bias) ----------------
// OUTM: 0 = bf16 [M][N], 1 = f32 [M][N], 2 = bf16 V^T as [B][H][HD][S]
#define GBM 128
#define GBN 64
#define GBK 64
#define GSTR 72   // padded LDS row stride (bf16 elems), row = 144B (16B-mult)

template <int ACT, int OUTM>
__global__ __launch_bounds__(128) void k_gemm(const u16* __restrict__ A,
                                              const u16* __restrict__ WT,
                                              const float* __restrict__ bias,
                                              u16* __restrict__ outB,
                                              float* __restrict__ outF,
                                              int M, int N, int K) {
  __shared__ alignas(16) u16 At[2][GBM * GSTR];   // 36.9 KB
  __shared__ alignas(16) u16 Bt[2][GBN * GSTR];   // 18.4 KB
  const int tid = threadIdx.x, lane = tid & 31, wave = tid >> 5;
  const int bm = blockIdx.y * GBM, bn = blockIdx.x * GBN;

  // async-stage one (A,B) K-slab into LDS buffer `buf`
  auto issue = [&](int buf, int k0) {
    const u16* ga = A + (size_t)(bm + tid) * K + k0;          // A: row per thread
    uint32_t la = lds_off(&At[buf][tid * GSTR]);
#pragma unroll
    for (int c = 0; c < 8; ++c) async_b128(la + c * 16, ga + c * 8);
    const int r = tid >> 1, cb = (tid & 1) * 32;              // B: half-row per thread
    const u16* gw = WT + (size_t)(bn + r) * K + k0 + cb;
    uint32_t lb = lds_off(&Bt[buf][r * GSTR + cb]);
#pragma unroll
    for (int c = 0; c < 4; ++c) async_b128(lb + c * 16, gw + c * 8);
  };

  v8f acc[2][4] = {};
  const int KT = K / GBK;
  issue(0, 0);
  for (int t = 0; t < KT; ++t) {
    const int cur = t & 1;
    wait_async0();        // own prefetch landed
    __syncthreads();      // everyone's prefetch visible; prior reads retired
    if (t + 1 < KT) issue(cur ^ 1, (t + 1) * GBK);   // overlap next slab
#pragma unroll
    for (int ks = 0; ks < 2; ++ks) {
      v16bf a0 = load_frag(At[cur], GSTR, wave * 32 + 0, ks * 32, lane);
      v16bf a1 = load_frag(At[cur], GSTR, wave * 32 + 16, ks * 32, lane);
#pragma unroll
      for (int nt = 0; nt < 4; ++nt) {
        v16bf bfr = load_frag(Bt[cur], GSTR, nt * 16, ks * 32, lane);
        acc[0][nt] = __builtin_amdgcn_wmma_f32_16x16x32_bf16(
            false, a0, false, bfr, (short)0, acc[0][nt], false, false);
        acc[1][nt] = __builtin_amdgcn_wmma_f32_16x16x32_bf16(
            false, a1, false, bfr, (short)0, acc[1][nt], false, false);
      }
    }
  }

  // epilogue: C layout -> element (m = r + (lane/16)*8, n = lane%16)
  const int half8 = ((lane >> 4) & 1) * 8;
#pragma unroll
  for (int mt = 0; mt < 2; ++mt)
#pragma unroll
    for (int nt = 0; nt < 4; ++nt) {
      const int n = bn + nt * 16 + (lane & 15);
      const float bn_ = bias[n];
#pragma unroll
      for (int r = 0; r < 8; ++r) {
        const int m = bm + wave * 32 + mt * 16 + r + half8;
        float c = acc[mt][nt][r] + bn_;
        if constexpr (ACT == 1) c *= 0.08838834764831845f;        // HD^-0.5
        if constexpr (ACT == 2) c = c / (1.f + __expf(-c));       // SiLU
        if constexpr (OUTM == 0) {
          outB[(size_t)m * N + n] = f2bf(c);
        } else if constexpr (OUTM == 1) {
          outF[(size_t)m * N + n] = c;
        } else {  // V^T: [B][H][HD][S] bf16
          const int s = m & (S_ - 1), bidx = m >> 11;
          const int hh = n >> 7, hd = n & (HD_ - 1);
          outB[((size_t)((bidx * H_ + hh) * HD_ + hd) << 11) + s] = f2bf(c);
        }
      }
    }
}

// ---------------- retention + groupnorm + gating ----------------
#define RT 64
#define KSTR (HD_ + 8)   // 136: K tile rows are tokens, cols are hd
#define VSTR 72          // V^T tile rows are hd, cols are tokens
#define SSTR 72

__global__ __launch_bounds__(128) void k_retention(const u16* __restrict__ q,
                                                   const u16* __restrict__ k,
                                                   const u16* __restrict__ vt,
                                                   const u16* __restrict__ gate,
                                                   const float* __restrict__ gnw,
                                                   const float* __restrict__ gnb,
                                                   const float* __restrict__ dec,
                                                   u16* __restrict__ y1) {
  __shared__ alignas(16) u16 Kt[RT * KSTR];        // 17.4 KB
  __shared__ alignas(16) u16 Vt[HD_ * VSTR];       // 18.4 KB (V^T tile)
  __shared__ alignas(16) u16 St[4 * 16 * SSTR];    // per-wave S strips (9.2 KB)
  const int tid = threadIdx.x, lane = tid & 31, wave = tid >> 5;
  const int half8 = ((lane >> 4) & 1) * 8;
  const int ib = blockIdx.x, h = blockIdx.y, b = blockIdx.z;
  const float lam = dec[h];
  const float glam = __expf(lam);          // per-row decay ratio
  const int i0 = ib * RT;
  const int ig0 = i0 + wave * 16 + half8;  // first row this lane covers
  u16* Sw = St + wave * 16 * SSTR;

  // Q strip: 16 rows per wave, HD=128 -> 4 K-step A fragments in regs
  v16bf qf[4];
  {
    const u16* qbase = q + ((size_t)(b * S_ + i0 + wave * 16 + (lane & 15))) * E_ + h * HD_;
#pragma unroll
    for (int f = 0; f < 4; ++f) {
      union { v16bf v; uint4 u[2]; } fr;
      fr.u[0] = *(const uint4*)(qbase + f * 32 + half8);
      fr.u[1] = *(const uint4*)(qbase + f * 32 + 16 + half8);
      qf[f] = fr.v;
    }
  }

  v8f acc[8] = {};      // 16 x 128 attn accumulator (8 column tiles)
  float rs[8] = {};     // running row sums of decayed scores

  // async-stage K (token-major) and V^T (hd-major) tiles for block j0
  auto stage = [&](int j0) {
    const int r = tid >> 1, c0 = (tid & 1) * 64;
    const u16* gk = k + ((size_t)(b * S_ + j0 + r)) * E_ + h * HD_ + c0;
    uint32_t lk = lds_off(&Kt[r * KSTR + c0]);
#pragma unroll
    for (int c = 0; c < 8; ++c) async_b128(lk + c * 16, gk + c * 8);
    const u16* gv = vt + ((size_t)((b * H_ + h) * HD_ + tid) << 11) + j0;
    uint32_t lv = lds_off(&Vt[tid * VSTR]);
#pragma unroll
    for (int c = 0; c < 8; ++c) async_b128(lv + c * 16, gv + c * 8);
    wait_async0();
    __syncthreads();
  };

  // one 16x16 score tile: S = (Q K^T) * gamma^(i-j), row-sum + bf16 stage.
  // decay factored: exp(lam*(ig0+r-jg)) = exp(lam*(ig0-jg)) * glam^r
  auto qk_tile = [&](int j0, int nt, bool maskDiag) {
    v8f s = {};
#pragma unroll
    for (int f = 0; f < 4; ++f) {
      v16bf kf = load_frag(Kt, KSTR, nt * 16, f * 32, lane);
      s = __builtin_amdgcn_wmma_f32_16x16x32_bf16(
          false, qf[f], false, kf, (short)0, s, false, false);
    }
    const int jg = j0 + nt * 16 + (lane & 15);
    float d = __expf(lam * (float)(ig0 - jg));
#pragma unroll
    for (int r = 0; r < 8; ++r) {
      float val = s[r] * d;
      if (maskDiag && (ig0 + r < jg)) val = 0.f;
      rs[r] += val;
      Sw[(r + half8) * SSTR + nt * 16 + (lane & 15)] = f2bf(val);
      d *= glam;
    }
  };

  // attn += S @ V; V^T tile gives the B operand in the standard layout
  auto accum_sv = [&]() {
#pragma unroll
    for (int f2 = 0; f2 < 2; ++f2) {
      v16bf sf = load_frag(Sw, SSTR, 0, f2 * 32, lane);
#pragma unroll
      for (int nt2 = 0; nt2 < 8; ++nt2) {
        v16bf vf = load_frag(Vt, VSTR, nt2 * 16, f2 * 32, lane);
        acc[nt2] = __builtin_amdgcn_wmma_f32_16x16x32_bf16(
            false, sf, false, vf, (short)0, acc[nt2], false, false);
      }
    }
  };

  // interior (strictly lower-triangular) blocks: no causal masking needed
  for (int jb = 0; jb < ib; ++jb) {
    stage(jb * RT);
#pragma unroll
    for (int nt = 0; nt < 4; ++nt) qk_tile(jb * RT, nt, false);
    accum_sv();
    __syncthreads();   // reads retired before next stage's async writes
  }

  // diagonal block: tiles above the wave's strip are identically zero
  {
    stage(i0);
#pragma unroll
    for (int nt = 0; nt < 4; ++nt) {
      if (nt > wave) {        // strictly above diagonal: zero, skip WMMA
#pragma unroll
        for (int r = 0; r < 8; ++r)
          Sw[(r + half8) * SSTR + nt * 16 + (lane & 15)] = 0;
      } else if (nt == wave) { // straddles diagonal: masked
        qk_tile(i0, nt, true);
      } else {                 // strictly below: unmasked
        qk_tile(i0, nt, false);
      }
    }
    accum_sv();
  }

  // row scale = clamp(|sum_j ret_ij|, 1); reduce across the 16 lanes of a row
  float scale[8];
#pragma unroll
  for (int r = 0; r < 8; ++r) {
    float t = rs[r];
#pragma unroll
    for (int m = 1; m < 16; m <<= 1) t += __shfl_xor(t, m, 16);
    scale[r] = fmaxf(fabsf(t), 1.0f);
  }

  // apply scale, gather GroupNorm stats over the 128-wide head group
  float sm[8] = {}, sq[8] = {};
#pragma unroll
  for (int nt2 = 0; nt2 < 8; ++nt2)
#pragma unroll
    for (int r = 0; r < 8; ++r) {
      float c = acc[nt2][r] / scale[r];
      acc[nt2][r] = c;
      sm[r] += c; sq[r] += c * c;
    }
  float mu[8], inv[8];
#pragma unroll
  for (int r = 0; r < 8; ++r) {
    float a = sm[r], c2 = sq[r];
#pragma unroll
    for (int m = 1; m < 16; m <<= 1) {
      a  += __shfl_xor(a, m, 16);
      c2 += __shfl_xor(c2, m, 16);
    }
    const float m_ = a * (1.0f / HD_);
    const float var = c2 * (1.0f / HD_) - m_ * m_;
    mu[r] = m_;
    inv[r] = rsqrtf(var + 1e-5f);
  }

  // y1 = silu_gate * groupnorm(attn)   (gate already SiLU'd by k_gemm ACT=2)
#pragma unroll
  for (int nt2 = 0; nt2 < 8; ++nt2) {
    const int ch = h * HD_ + nt2 * 16 + (lane & 15);
    const float w = gnw[ch], bb = gnb[ch];
#pragma unroll
    for (int r = 0; r < 8; ++r) {
      const int ig = i0 + wave * 16 + r + half8;
      const size_t off = (size_t)(b * S_ + ig) * E_ + ch;
      const float g = (acc[nt2][r] - mu[r]) * inv[r] * w + bb;
      y1[off] = f2bf(bf2f(gate[off]) * g);
    }
  }
}

extern "C" void kernel_launch(void* const* d_in, const int* in_sizes, int n_in,
                              void* d_out, int out_size, void* d_ws, size_t ws_size,
                              hipStream_t stream) {
  (void)in_sizes; (void)n_in; (void)out_size; (void)ws_size;
  const float* x   = (const float*)d_in[0];
  const float* Wq  = (const float*)d_in[1];
  const float* bq  = (const float*)d_in[2];
  const float* Wk  = (const float*)d_in[3];
  const float* bk  = (const float*)d_in[4];
  const float* Wv  = (const float*)d_in[5];
  const float* bv  = (const float*)d_in[6];
  const float* Wg  = (const float*)d_in[7];
  const float* bg  = (const float*)d_in[8];
  const float* Wo  = (const float*)d_in[9];
  const float* bo  = (const float*)d_in[10];
  const float* gnw = (const float*)d_in[11];
  const float* gnb = (const float*)d_in[12];
  const float* dec = (const float*)d_in[13];
  float* out = (float*)d_out;

  const size_t M = (size_t)B_ * S_;          // 4096 tokens
  const size_t act_elems = M * E_;           // 8,388,608
  const size_t w_elems = (size_t)E_ * E_;    // 4,194,304

  char* ws = (char*)d_ws;
  u16* xb  = (u16*)ws; ws += act_elems * 2;
  u16* qb  = (u16*)ws; ws += act_elems * 2;
  u16* kb  = (u16*)ws; ws += act_elems * 2;
  u16* vb  = (u16*)ws; ws += act_elems * 2;   // holds V^T [B][H][HD][S]
  u16* gb  = (u16*)ws; ws += act_elems * 2;
  u16* WqT = (u16*)ws; ws += w_elems * 2;
  u16* WkT = (u16*)ws; ws += w_elems * 2;
  u16* WvT = (u16*)ws; ws += w_elems * 2;
  u16* WgT = (u16*)ws; ws += w_elems * 2;
  u16* WoT = (u16*)ws; ws += w_elems * 2;
  u16* y1  = xb;   // x consumed by projection GEMMs before retention writes y1

  k_cvt<<<dim3((unsigned)(act_elems / 4 / 256)), dim3(256), 0, stream>>>(x, xb, (int)(act_elems / 4));
  dim3 tgrid(E_ / 32, E_ / 32), tblk(32, 8);
  k_cvt_t<<<tgrid, tblk, 0, stream>>>(Wq, WqT);
  k_cvt_t<<<tgrid, tblk, 0, stream>>>(Wk, WkT);
  k_cvt_t<<<tgrid, tblk, 0, stream>>>(Wv, WvT);
  k_cvt_t<<<tgrid, tblk, 0, stream>>>(Wg, WgT);
  k_cvt_t<<<tgrid, tblk, 0, stream>>>(Wo, WoT);

  dim3 ggrid(E_ / GBN, (unsigned)(M / GBM));          // (32, 32)
  k_gemm<1, 0><<<ggrid, 128, 0, stream>>>(xb, WqT, bq, qb, nullptr, (int)M, E_, E_);
  k_gemm<0, 0><<<ggrid, 128, 0, stream>>>(xb, WkT, bk, kb, nullptr, (int)M, E_, E_);
  k_gemm<0, 2><<<ggrid, 128, 0, stream>>>(xb, WvT, bv, vb, nullptr, (int)M, E_, E_);
  k_gemm<2, 0><<<ggrid, 128, 0, stream>>>(xb, WgT, bg, gb, nullptr, (int)M, E_, E_);

  k_retention<<<dim3(S_ / RT, H_, B_), 128, 0, stream>>>(qb, kb, vb, gb, gnw, gnb, dec, y1);

  k_gemm<0, 1><<<ggrid, 128, 0, stream>>>(y1, WoT, bo, nullptr, out, (int)M, E_, E_);
}
